// GNNQLearning_47055661695332
// MI455X (gfx1250) — matrix operand order repaired and checked by reading, not compiled, not run on hardware
//
#include <hip/hip_runtime.h>

// ---------------- types ----------------
typedef __attribute__((ext_vector_type(16))) _Float16 v16h;
typedef __attribute__((ext_vector_type(8)))  _Float16 v8h;
typedef __attribute__((ext_vector_type(8)))  float    v8f;

static inline unsigned cdiv(unsigned a, unsigned b) { return (a + b - 1) / b; }
static inline size_t align256(size_t x) { return (x + 255) & ~(size_t)255; }

// ---------------- degree / norm ----------------
__global__ void k_fill_f32(float* __restrict__ p, float v, int n) {
    int i = blockIdx.x * blockDim.x + threadIdx.x;
    if (i < n) p[i] = v;
}

__global__ void k_deg_edges(const long long* __restrict__ dst, float* __restrict__ deg, int E) {
    int e = blockIdx.x * blockDim.x + threadIdx.x;
    if (e < E) atomicAdd(&deg[(int)dst[e]], 1.0f);
}

__global__ void k_dinv(float* __restrict__ deg, int n) {
    int i = blockIdx.x * blockDim.x + threadIdx.x;
    if (i < n) deg[i] = rsqrtf(deg[i]);   // deg >= 1 (self loop), always valid
}

// ---------------- edge precompute: int32 pairs + fused normalization -------
__global__ void k_prep_edges(const long long* __restrict__ ei, const float* __restrict__ dinv,
                             int2* __restrict__ sd, float* __restrict__ coef, int E) {
    int e = blockIdx.x * blockDim.x + threadIdx.x;
    if (e < E) {
        int s = (int)ei[e];
        int d = (int)ei[E + e];
        sd[e]   = make_int2(s, d);
        coef[e] = dinv[s] * dinv[d];
    }
}

// ---------------- layer 1: fin=1 outer product ----------------
__global__ void k_l1_transform(const float* __restrict__ x, const float* __restrict__ W1,
                               float* __restrict__ H, int n) {
    int i = blockIdx.x * blockDim.x + threadIdx.x;
    if (i < n * 64) H[i] = x[i >> 6] * W1[i & 63];
}

// ---------------- aggregation: bias + self-loop init ----------------
template <int F, int LOGF>
__global__ void k_agg_init(const float* __restrict__ H, const float* __restrict__ dinv,
                           const float* __restrict__ b, float* __restrict__ OUT, int n) {
    int i = blockIdx.x * blockDim.x + threadIdx.x;
    if (i < n * F) {
        int node = i >> LOGF, f = i & (F - 1);
        float di = dinv[node];
        OUT[i] = b[f] + di * di * H[i];
    }
}

// ------------- aggregation: edge scatter, precomputed coef (fast path) ------
template <int F>
__global__ void k_agg_edges_pre(const int2* __restrict__ sd, const float* __restrict__ coef,
                                const float* __restrict__ H, float* __restrict__ OUT, int E) {
    const int C = F / 4;
    int t = blockIdx.x * blockDim.x + threadIdx.x;
    if (t >= E * C) return;
    int e = t / C, c = t - e * C;
    int2  p  = sd[e];
    float cf = coef[e];
    const float4 hv = *(const float4*)(H + (size_t)p.x * F + c * 4);
    float* o = OUT + (size_t)p.y * F + c * 4;
    atomicAdd(o + 0, cf * hv.x);
    atomicAdd(o + 1, cf * hv.y);
    atomicAdd(o + 2, cf * hv.z);
    atomicAdd(o + 3, cf * hv.w);
}

// ------------- aggregation: edge scatter, direct (fallback, small ws) -------
template <int F>
__global__ void k_agg_edges(const long long* __restrict__ ei, const float* __restrict__ H,
                            const float* __restrict__ dinv, float* __restrict__ OUT, int E) {
    const int C = F / 4;
    int t = blockIdx.x * blockDim.x + threadIdx.x;
    if (t >= E * C) return;
    int e = t / C, c = t - e * C;
    int s = (int)ei[e];
    int d = (int)ei[E + e];
    float cf = dinv[s] * dinv[d];
    const float4 hv = *(const float4*)(H + (size_t)s * F + c * 4);
    float* o = OUT + (size_t)d * F + c * 4;
    atomicAdd(o + 0, cf * hv.x);
    atomicAdd(o + 1, cf * hv.y);
    atomicAdd(o + 2, cf * hv.z);
    atomicAdd(o + 3, cf * hv.w);
}

// ---------------- relu in place + f16 staging for next GEMM ----------------
template <int F>
__global__ void k_relu_half(float* __restrict__ OUT, _Float16* __restrict__ Xh, int n) {
    int i = blockIdx.x * blockDim.x + threadIdx.x;
    if (i < n * F) {
        float v = OUT[i];
        v = v > 0.0f ? v : 0.0f;
        OUT[i] = v;
        Xh[i]  = (_Float16)v;
    }
}

// ---------------- W [K, Nf] f32 -> Wt [Nf, K] f16 (column major for B frags) ----
__global__ void k_w_half_t(const float* __restrict__ W, _Float16* __restrict__ Wt, int K, int Nf) {
    int i = blockIdx.x * blockDim.x + threadIdx.x;
    if (i < K * Nf) {
        int nn = i / K, k = i - nn * K;
        Wt[i] = (_Float16)W[k * Nf + nn];
    }
}

// ---------------- WMMA GEMM: H[n,NF] = Xh[n,K] @ W[K,NF]  (f16 in, f32 acc) ----
// One wave computes one 16x16 output tile; block = 8 waves = 8 consecutive M tiles.
template <int K, int NF>
__global__ void k_gemm_wmma(const _Float16* __restrict__ Xh, const _Float16* __restrict__ Wt,
                            float* __restrict__ H, int n) {
    const int lane   = threadIdx.x & 31;
    const int wave   = threadIdx.x >> 5;
    const int mtile  = blockIdx.x * (blockDim.x >> 5) + wave;
    const int mtiles = n >> 4;                 // n is a multiple of 16
    if (mtile >= mtiles) return;               // wave-uniform: EXEC stays all-ones

    const int g    = lane >> 4;                // lane half-group
    const int l15  = lane & 15;
    const int row  = (mtile << 4) + l15;       // A row for this lane
    const int ncol = (blockIdx.y << 4) + l15;  // B/D column for this lane

    v8f acc = {};
#pragma unroll
    for (int k0 = 0; k0 < K; k0 += 32) {
        // A fragment: halves j<8 -> K=k0+g*8+j ; j>=8 -> K=k0+16+g*8+(j-8)
        const _Float16* ap = Xh + (size_t)row * K + k0 + g * 8;
        v8h alo = *(const v8h*)(ap);
        v8h ahi = *(const v8h*)(ap + 16);
        v16h a;
#pragma unroll
        for (int j = 0; j < 8; ++j) { a[j] = alo[j]; a[j + 8] = ahi[j]; }

        // B fragment: halves j<8 -> K=k0+g*8+j ; j>=8 -> K=k0+16+g*8+(j-8) at fixed column
        const _Float16* bp = Wt + (size_t)ncol * K + k0 + g * 8;
        v8h blo = *(const v8h*)(bp);
        v8h bhi = *(const v8h*)(bp + 16);
        v16h bm;
#pragma unroll
        for (int j = 0; j < 8; ++j) { bm[j] = blo[j]; bm[j + 8] = bhi[j]; }

        acc = __builtin_amdgcn_wmma_f32_16x16x32_f16(false, a, false, bm,
                                                     (short)0, acc, false, false);
    }

    // D layout: VGPR r -> M = r + 8*g, N = lane&15
    const int mbase = (mtile << 4) + g * 8;
#pragma unroll
    for (int r = 0; r < 8; ++r)
        H[(size_t)(mbase + r) * NF + ncol] = acc[r];
}

// ---------------- layer 4: fout=1 dot product ----------------
__global__ void k_l4_dot(const float* __restrict__ X, const float* __restrict__ W4,
                         float* __restrict__ Hd, int n) {
    int i = blockIdx.x * blockDim.x + threadIdx.x;
    if (i < n) {
        float s = 0.0f;
#pragma unroll
        for (int k = 0; k < 16; ++k) s += X[i * 16 + k] * W4[k];
        Hd[i] = s;
    }
}

__global__ void k_out_init(const float* __restrict__ Hd, const float* __restrict__ dinv,
                           const float* __restrict__ b4, float* __restrict__ out, int n) {
    int i = blockIdx.x * blockDim.x + threadIdx.x;
    if (i < n) {
        float di = dinv[i];
        out[i] = b4[0] + di * di * Hd[i];
    }
}

__global__ void k_out_edges_pre(const int2* __restrict__ sd, const float* __restrict__ coef,
                                const float* __restrict__ Hd, float* __restrict__ out, int E) {
    int e = blockIdx.x * blockDim.x + threadIdx.x;
    if (e < E) {
        int2 p = sd[e];
        atomicAdd(&out[p.y], coef[e] * Hd[p.x]);
    }
}

__global__ void k_out_edges(const long long* __restrict__ ei, const float* __restrict__ Hd,
                            const float* __restrict__ dinv, float* __restrict__ out, int E) {
    int e = blockIdx.x * blockDim.x + threadIdx.x;
    if (e < E) {
        int s = (int)ei[e];
        int d = (int)ei[E + e];
        atomicAdd(&out[d], dinv[s] * dinv[d] * Hd[s]);
    }
}

// ---------------- launch ----------------
extern "C" void kernel_launch(void* const* d_in, const int* in_sizes, int n_in,
                              void* d_out, int out_size, void* d_ws, size_t ws_size,
                              hipStream_t stream) {
    const float*     x  = (const float*)d_in[0];
    const long long* ei = (const long long*)d_in[1];   // int64 [2, E]
    const float* W1 = (const float*)d_in[2];
    const float* b1 = (const float*)d_in[3];
    const float* W2 = (const float*)d_in[4];
    const float* b2 = (const float*)d_in[5];
    const float* W3 = (const float*)d_in[6];
    const float* b3 = (const float*)d_in[7];
    const float* W4 = (const float*)d_in[8];
    const float* b4 = (const float*)d_in[9];
    float* out = (float*)d_out;

    const int n = in_sizes[0];          // x is [N, 1]
    const int E = in_sizes[1] / 2;

    // scratch layout
    char* ws = (char*)d_ws;
    size_t off = 0;
    float* dinv = (float*)(ws + off);           off += align256((size_t)n * 4);
    float* H    = (float*)(ws + off);           off += align256((size_t)n * 64 * 4);
    _Float16* Xh = (_Float16*)(ws + off);       off += align256((size_t)n * 64 * 2);
    float* OUT  = (float*)(ws + off);           off += align256((size_t)n * 64 * 4);
    _Float16* W2t = (_Float16*)(ws + off);
    _Float16* W3t = W2t + 64 * 32;              off += align256((size_t)(64 * 32 + 32 * 16) * 2);
    int2*  sd   = (int2*)(ws + off);            off += align256((size_t)E * 8);
    float* coef = (float*)(ws + off);           off += align256((size_t)E * 4);
    const bool use_pre = (ws_size >= off);      // edge-precompute fast path fits?

    const int T = 256;

    // degree + symmetric norm (self loop counted via init = 1)
    k_fill_f32<<<cdiv(n, T), T, 0, stream>>>(dinv, 1.0f, n);
    k_deg_edges<<<cdiv(E, T), T, 0, stream>>>(ei + E, dinv, E);
    k_dinv<<<cdiv(n, T), T, 0, stream>>>(dinv, n);
    if (use_pre)
        k_prep_edges<<<cdiv(E, T), T, 0, stream>>>(ei, dinv, sd, coef, E);

    // weight staging for WMMA (f16, column major)
    k_w_half_t<<<cdiv(64 * 32, T), T, 0, stream>>>(W2, W2t, 64, 32);
    k_w_half_t<<<cdiv(32 * 16, T), T, 0, stream>>>(W3, W3t, 32, 16);

    // ---- layer 1: 1 -> 64 ----
    k_l1_transform<<<cdiv(n * 64, T), T, 0, stream>>>(x, W1, H, n);
    k_agg_init<64, 6><<<cdiv(n * 64, T), T, 0, stream>>>(H, dinv, b1, OUT, n);
    if (use_pre) k_agg_edges_pre<64><<<cdiv(E * 16, T), T, 0, stream>>>(sd, coef, H, OUT, E);
    else         k_agg_edges<64><<<cdiv(E * 16, T), T, 0, stream>>>(ei, H, dinv, OUT, E);
    k_relu_half<64><<<cdiv(n * 64, T), T, 0, stream>>>(OUT, Xh, n);

    // ---- layer 2: 64 -> 32 (WMMA) ----
    {
        dim3 grid(cdiv(n / 16, 8), 32 / 16);
        k_gemm_wmma<64, 32><<<grid, 256, 0, stream>>>(Xh, W2t, H, n);
    }
    k_agg_init<32, 5><<<cdiv(n * 32, T), T, 0, stream>>>(H, dinv, b2, OUT, n);
    if (use_pre) k_agg_edges_pre<32><<<cdiv(E * 8, T), T, 0, stream>>>(sd, coef, H, OUT, E);
    else         k_agg_edges<32><<<cdiv(E * 8, T), T, 0, stream>>>(ei, H, dinv, OUT, E);
    k_relu_half<32><<<cdiv(n * 32, T), T, 0, stream>>>(OUT, Xh, n);

    // ---- layer 3: 32 -> 16 (WMMA) ----
    {
        dim3 grid(cdiv(n / 16, 8), 1);
        k_gemm_wmma<32, 16><<<grid, 256, 0, stream>>>(Xh, W3t, H, n);
    }
    k_agg_init<16, 4><<<cdiv(n * 16, T), T, 0, stream>>>(H, dinv, b3, OUT, n);
    if (use_pre) k_agg_edges_pre<16><<<cdiv(E * 4, T), T, 0, stream>>>(sd, coef, H, OUT, E);
    else         k_agg_edges<16><<<cdiv(E * 4, T), T, 0, stream>>>(ei, H, dinv, OUT, E);
    k_relu_half<16><<<cdiv(n * 16, T), T, 0, stream>>>(OUT, Xh, n);

    // ---- layer 4: 16 -> 1 ----
    k_l4_dot<<<cdiv(n, T), T, 0, stream>>>(OUT, W4, H, n);
    k_out_init<<<cdiv(n, T), T, 0, stream>>>(H, dinv, b4, out, n);
    if (use_pre) k_out_edges_pre<<<cdiv(E, T), T, 0, stream>>>(sd, coef, H, out, E);
    else         k_out_edges<<<cdiv(E, T), T, 0, stream>>>(ei, H, dinv, out, E);
}
